// SNNLLoss_60155311948301
// MI455X (gfx1250) — matrix operand order repaired and checked by reading, not compiled, not run on hardware
//
#include <hip/hip_runtime.h>

typedef __attribute__((ext_vector_type(16))) __bf16        v16bf;
typedef __attribute__((ext_vector_type(8)))  float         v8f;
typedef __attribute__((ext_vector_type(4)))  unsigned int  u32x4;

#define DDIM   128
#define EPSF   1e-8f
#define JSPLIT 4

// ---------------------------------------------------------------------------
// Kernel 1: zero accumulators, compute per-row squared norms, and split-convert
// embeddings to bf16 hi/lo planes (round-to-nearest-even) stored row-major.
// ---------------------------------------------------------------------------
__global__ __launch_bounds__(256) void snnl_prep_kernel(
    const float* __restrict__ emb,
    unsigned short* __restrict__ ehi,
    unsigned short* __restrict__ elo,
    float* __restrict__ sq,
    float* __restrict__ numAcc,
    float* __restrict__ denAcc,
    float* __restrict__ cntAcc,
    float* __restrict__ scal,
    int N)
{
    const int row = blockIdx.x * blockDim.x + threadIdx.x;
    if (row >= N) return;
    if (row == 0) { scal[0] = 0.0f; scal[1] = 0.0f; }
    numAcc[row] = 0.0f; denAcc[row] = 0.0f; cntAcc[row] = 0.0f;

    const float* src = emb + (size_t)row * DDIM;
    unsigned short* hdst = ehi + (size_t)row * DDIM;
    unsigned short* ldst = elo + (size_t)row * DDIM;
    float s = 0.0f;
    #pragma unroll 8
    for (int k = 0; k < DDIM; ++k) {
        const float x = src[k];
        s += x * x;
        unsigned u = __builtin_bit_cast(unsigned int, x);
        unsigned short h = (unsigned short)((u + (0x7FFFu + ((u >> 16) & 1u))) >> 16);
        const float hf = __builtin_bit_cast(float, (unsigned)((unsigned)h << 16));
        const float l = x - hf;
        unsigned ul = __builtin_bit_cast(unsigned int, l);
        unsigned short hl = (unsigned short)((ul + (0x7FFFu + ((ul >> 16) & 1u))) >> 16);
        hdst[k] = h;
        ldst[k] = hl;
    }
    sq[row] = s;
}

// ---------------------------------------------------------------------------
// Kernel 2: fused Gram(WMMA bf16 split) + exp + masked row accumulation.
// One wave = one 16-row i-block x one quarter of the j range.
// Three independent accumulator chains (hi*hi, hi*lo, lo*hi) keep the XDL
// pipe full instead of one serial 12-deep accumulation chain.
// ---------------------------------------------------------------------------
__global__ __launch_bounds__(256) void snnl_gemm_kernel(
    const unsigned short* __restrict__ ehi,
    const unsigned short* __restrict__ elo,
    const float* __restrict__ sq,
    const int* __restrict__ labels,
    float* __restrict__ numAcc,
    float* __restrict__ denAcc,
    float* __restrict__ cntAcc,
    int N)
{
    const int lane    = threadIdx.x & 31;
    const int waveId  = blockIdx.x * (blockDim.x >> 5) + (threadIdx.x >> 5);
    const int iBlock  = waveId / JSPLIT;
    const int jSlice  = waveId % JSPLIT;
    const int jTiles  = (N >> 4) / JSPLIT;
    const int tBeg    = jSlice * jTiles;
    const int tEnd    = tBeg + jTiles;

    const int i0      = iBlock << 4;
    const int halfSel = lane >> 4;   // 0: lanes 0-15, 1: lanes 16-31
    const int lrow    = lane & 15;

    union VReg { v16bf bf; u32x4 q[2]; };

    // ---- A operand (16 rows x K=128), hi & lo planes, WMMA A layout:
    // lane half g: VGPR0-3 hold K = 32c + 8g + {0..7}, VGPR4-7 hold K+16.
    VReg aHi[4], aLo[4];
    {
        const size_t rbase = (size_t)(i0 + lrow) * DDIM;
        #pragma unroll
        for (int c = 0; c < 4; ++c) {
            const int kb = 32 * c + halfSel * 8;
            aHi[c].q[0] = *(const u32x4*)(ehi + rbase + kb);
            aHi[c].q[1] = *(const u32x4*)(ehi + rbase + kb + 16);
            aLo[c].q[0] = *(const u32x4*)(elo + rbase + kb);
            aLo[c].q[1] = *(const u32x4*)(elo + rbase + kb + 16);
        }
    }

    // C-layout row invariants: VGPR r of the accumulator holds row i0+r (+8 for hi half).
    float sqi[8]; int labi[8]; int irow[8];
    #pragma unroll
    for (int r = 0; r < 8; ++r) {
        irow[r] = i0 + r + halfSel * 8;
        sqi[r]  = sq[irow[r]];
        labi[r] = labels[irow[r]];
    }

    float numv[8], denv[8], cntv[8];
    #pragma unroll
    for (int r = 0; r < 8; ++r) { numv[r] = 0.0f; denv[r] = 0.0f; cntv[r] = 0.0f; }

    for (int t = tBeg; t < tEnd; ++t) {
        const int jrow = (t << 4) + lrow;
        const size_t jb = (size_t)jrow * DDIM;

        // ---- B operand (K=32 x 16 cols), layout: lane = column N, lane half g
        // holds K = 32c + 8g + {0..7} in VGPR0-3, K+16 in VGPR4-7.
        VReg bHi[4], bLo[4];
        #pragma unroll
        for (int c = 0; c < 4; ++c) {
            const int kb = 32 * c + halfSel * 8;
            bHi[c].q[0] = *(const u32x4*)(ehi + jb + kb);
            bHi[c].q[1] = *(const u32x4*)(ehi + jb + kb + 16);
            bLo[c].q[0] = *(const u32x4*)(elo + jb + kb);
            bLo[c].q[1] = *(const u32x4*)(elo + jb + kb + 16);
        }
        const float sqj  = sq[jrow];
        const int   labj = labels[jrow];

        if (t + 1 < tEnd) {  // wave-uniform; pull next tile's rows toward L0
            __builtin_prefetch(ehi + jb + DDIM * 16, 0, 1);
            __builtin_prefetch(elo + jb + DDIM * 16, 0, 1);
        }

        // Split-bf16 compensated GEMM: hi*hi + hi*lo + lo*hi, f32 accumulate,
        // three independent XDL accumulation chains of depth 4 each.
        v8f accHH = {};
        v8f accHL = {};
        v8f accLH = {};
        #pragma unroll
        for (int c = 0; c < 4; ++c) {
            accHH = __builtin_amdgcn_wmma_f32_16x16x32_bf16(false, aHi[c].bf, false, bHi[c].bf, (short)0, accHH, false, false);
            accHL = __builtin_amdgcn_wmma_f32_16x16x32_bf16(false, aHi[c].bf, false, bLo[c].bf, (short)0, accHL, false, false);
            accLH = __builtin_amdgcn_wmma_f32_16x16x32_bf16(false, aLo[c].bf, false, bHi[c].bf, (short)0, accLH, false, false);
        }

        // Fused epilogue on the C layout (lane's column j is the same for all 8 rows)
        #pragma unroll
        for (int r = 0; r < 8; ++r) {
            const float g = (accHH[r] + accHL[r]) + accLH[r];
            float d2 = __builtin_fmaf(-2.0f, g, sqi[r] + sqj);
            d2 = fmaxf(d2, 0.0f);
            const float e  = __expf(-d2);
            const bool ns = (jrow != irow[r]);
            const bool sm = ns && (labj == labi[r]);
            denv[r] += ns ? e : 0.0f;
            numv[r] += sm ? e : 0.0f;
            cntv[r] += sm ? 1.0f : 0.0f;
        }
    }

    // Row sums live across one 16-lane half: butterfly-reduce, then one atomic
    // per row from lane 0 of each half (lanes 0 and 16).
    #pragma unroll
    for (int r = 0; r < 8; ++r) {
        float nu = numv[r], de = denv[r], ct = cntv[r];
        #pragma unroll
        for (int m = 8; m >= 1; m >>= 1) {
            nu += __shfl_xor(nu, m, 32);
            de += __shfl_xor(de, m, 32);
            ct += __shfl_xor(ct, m, 32);
        }
        if (lrow == 0) {
            atomicAdd(&numAcc[irow[r]], nu);
            atomicAdd(&denAcc[irow[r]], de);
            atomicAdd(&cntAcc[irow[r]], ct);
        }
    }
}

// ---------------------------------------------------------------------------
// Kernel 3: per-row loss + block reduction into scalar accumulators.
// ---------------------------------------------------------------------------
__global__ __launch_bounds__(256) void snnl_rowloss_kernel(
    const float* __restrict__ numAcc,
    const float* __restrict__ denAcc,
    const float* __restrict__ cntAcc,
    float* __restrict__ scal,
    int N)
{
    __shared__ float sSum[256];
    __shared__ float sCnt[256];
    const int i = blockIdx.x * blockDim.x + threadIdx.x;
    float contrib = 0.0f, c = 0.0f;
    if (i < N && cntAcc[i] > 0.5f) {
        contrib = __logf(denAcc[i] + EPSF) - __logf(numAcc[i] + EPSF);
        c = 1.0f;
    }
    sSum[threadIdx.x] = contrib;
    sCnt[threadIdx.x] = c;
    __syncthreads();
    for (int s = 128; s > 0; s >>= 1) {
        if ((int)threadIdx.x < s) {
            sSum[threadIdx.x] += sSum[threadIdx.x + s];
            sCnt[threadIdx.x] += sCnt[threadIdx.x + s];
        }
        __syncthreads();
    }
    if (threadIdx.x == 0) {
        atomicAdd(&scal[0], sSum[0]);
        atomicAdd(&scal[1], sCnt[0]);
    }
}

__global__ void snnl_final_kernel(const float* __restrict__ scal, float* __restrict__ out)
{
    out[0] = scal[0] / fmaxf(scal[1], 1.0f);
}

// ---------------------------------------------------------------------------
extern "C" void kernel_launch(void* const* d_in, const int* in_sizes, int n_in,
                              void* d_out, int out_size, void* d_ws, size_t ws_size,
                              hipStream_t stream)
{
    const float* emb    = (const float*)d_in[0];
    const int*   labels = (const int*)d_in[1];
    const int N = in_sizes[1];              // 8192 (labels count); D fixed at 128

    // Workspace layout (256B-aligned sections):
    //  [0..1]   scalar loss-sum / valid-count
    //  +64      num[N], den[N], cnt[N], sq[N]   (f32)
    //  then     E_hi[N*D], E_lo[N*D]            (bf16 planes)
    float* W       = (float*)d_ws;
    float* scal    = W;
    float* numAcc  = W + 64;
    float* denAcc  = numAcc + N;
    float* cntAcc  = denAcc + N;
    float* sq      = cntAcc + N;
    unsigned short* ehi = (unsigned short*)(sq + N);
    unsigned short* elo = ehi + (size_t)N * DDIM;

    snnl_prep_kernel<<<(N + 255) / 256, 256, 0, stream>>>(
        emb, ehi, elo, sq, numAcc, denAcc, cntAcc, scal, N);

    const int waves  = (N >> 4) * JSPLIT;   // 2048 waves
    const int blocks = waves / 8;           // 8 waves (256 threads) per block
    snnl_gemm_kernel<<<blocks, 256, 0, stream>>>(
        ehi, elo, sq, labels, numAcc, denAcc, cntAcc, N);

    snnl_rowloss_kernel<<<(N + 255) / 256, 256, 0, stream>>>(
        numAcc, denAcc, cntAcc, scal, N);

    snnl_final_kernel<<<1, 1, 0, stream>>>(scal, (float*)d_out);
}
